// LEXGAT_57621281243612
// MI455X (gfx1250) — matrix operand order repaired and checked by reading, not compiled, not run on hardware
//
#include <hip/hip_runtime.h>
#include <hip/hip_bf16.h>

typedef __attribute__((ext_vector_type(16))) _Float16 v16h;
typedef __attribute__((ext_vector_type(8)))  _Float16 v8h;
typedef __attribute__((ext_vector_type(8)))  float    v8f;

#define NNODES 50000
#define NEDGES 400000
#define SLOPE  0.2f

// ---------- WMMA fragment helpers (contiguous k order, same in A and B) ----------
// A (16x32 f16): lane holds row M=lane%16; K elements: lanes<16 -> K 0..15,
// lanes>=16 -> K 16..31. Source is row-major f16 with leading dim `ld`.
__device__ __forceinline__ v16h load_fragA(const _Float16* base, int ld, int lane) {
    const _Float16* q = base + (lane & 15) * ld + ((lane & 16) ? 16 : 0);
    v8h lo = *(const v8h*)q;
    v8h hi = *(const v8h*)(q + 8);
    v16h f;
#pragma unroll
    for (int i = 0; i < 8; ++i) { f[i] = lo[i]; f[i + 8] = hi[i]; }
    return f;
}
// B (32x16 f16): lane holds column n=lane%16; lanes<16 -> K 0..15, lanes>=16 -> K 16..31,
// contiguous. We feed W row-major (row = output column n), so this is a contiguous 32B load.
__device__ __forceinline__ v16h load_fragB(const _Float16* base, int ld, int lane) {
    const _Float16* q = base + (lane & 15) * ld + ((lane & 16) ? 16 : 0);
    return *(const v16h*)q;
}
__device__ __forceinline__ v8f wmma_f16(v16h a, v16h b, v8f c) {
    return __builtin_amdgcn_wmma_f32_16x16x32_f16(false, a, false, b, (short)0, c, false, false);
}

// Monotonic float -> uint key for atomicMax-based segment max.
__device__ __forceinline__ unsigned fkey(float x) {
    unsigned b = __float_as_uint(x);
    return (b & 0x80000000u) ? ~b : (b | 0x80000000u);
}
__device__ __forceinline__ float fdec(unsigned k) {
    unsigned b = (k & 0x80000000u) ? (k ^ 0x80000000u) : ~k;
    return __uint_as_float(b);
}

// ---------------------------- weight f32 -> f16 ----------------------------
__global__ void cvt_f16_kernel(const float* __restrict__ s, _Float16* __restrict__ d, int n) {
    int i = blockIdx.x * blockDim.x + threadIdx.x;
    if (i < n) d[i] = (_Float16)s[i];
}

// ------------- node pipeline: z -> Ws0/Ws1 blend -> fc -> feat -------------
__global__ __launch_bounds__(256) void node_feat_kernel(
    const float* __restrict__ x, const float* __restrict__ y, const float* __restrict__ emb,
    const float* __restrict__ Ws0_b, const float* __restrict__ Ws1_b,
    const _Float16* __restrict__ Ws0h, const _Float16* __restrict__ Ws1h,
    const _Float16* __restrict__ fch, float* __restrict__ feat, int Ncnt)
{
    __shared__ _Float16 sA[128][136];
    __shared__ _Float16 sZ[128][136];
    __shared__ float    sP[128];

    const int n0  = blockIdx.x * 128;
    const int tid = threadIdx.x;

    // Phase 1: z = x + (1-p)*emb0 + p*emb1  (f16 into LDS; OOB rows -> 0)
    for (int idx = tid; idx < 128 * 128; idx += 256) {
        int r = idx >> 7, c = idx & 127;
        int n = n0 + r;
        float p = (n < Ncnt) ? y[n] : 0.f;
        if (c == 0) sP[r] = p;
        float zn = (1.f - p) * emb[c] + p * emb[128 + c];
        float v  = (n < Ncnt) ? (x[(size_t)n * 128 + c] + zn) : 0.f;
        sA[r][c] = (_Float16)v;
    }
    __syncthreads();

    const int wave = tid >> 5, lane = tid & 31;
    const int r0   = wave * 16;
    const int rlo  = r0 + ((lane & 16) ? 8 : 0);   // block-local row of C vgpr 0
    const int cl   = lane & 15;

    float pv[8];
#pragma unroll
    for (int r = 0; r < 8; ++r) pv[r] = sP[rlo + r];

    // Phase 2: z2 = (1-p)*(z@Ws0^T+b0) + p*(z@Ws1^T+b1) -> sZ (f16)
    for (int c = 0; c < 8; ++c) {
        v8f a0 = {}, a1 = {};
#pragma unroll
        for (int k = 0; k < 4; ++k) {
            v16h af = load_fragA(&sA[r0][k * 32], 136, lane);
            v16h b0 = load_fragB(Ws0h + (size_t)(c * 16) * 128 + k * 32, 128, lane);
            v16h b1 = load_fragB(Ws1h + (size_t)(c * 16) * 128 + k * 32, 128, lane);
            a0 = wmma_f16(af, b0, a0);
            a1 = wmma_f16(af, b1, a1);
        }
        int cn = c * 16 + cl;
        float bb0 = Ws0_b[cn], bb1 = Ws1_b[cn];
#pragma unroll
        for (int r = 0; r < 8; ++r) {
            float p = pv[r];
            sZ[rlo + r][cn] = (_Float16)((1.f - p) * (a0[r] + bb0) + p * (a1[r] + bb1));
        }
    }
    // Each wave reads back only the sZ rows it wrote (in-order LDS per wave).

    // Phase 3: feat = z2 @ fc^T  (512 cols)
    for (int c = 0; c < 32; ++c) {
        v8f acc = {};
#pragma unroll
        for (int k = 0; k < 4; ++k) {
            v16h af = load_fragA(&sZ[r0][k * 32], 136, lane);
            v16h bf = load_fragB(fch + (size_t)(c * 16) * 128 + k * 32, 128, lane);
            acc = wmma_f16(af, bf, acc);
        }
        int cn = c * 16 + cl;
#pragma unroll
        for (int r = 0; r < 8; ++r) {
            int n = n0 + rlo + r;
            if (n < Ncnt) feat[(size_t)n * 512 + cn] = acc[r];
        }
    }
}

// ------------------------- el/er: wave per (n, head) -------------------------
__global__ __launch_bounds__(256) void attn_dot_kernel(
    const float* __restrict__ feat, const float* __restrict__ attn_l,
    const float* __restrict__ attn_r, float* __restrict__ el, float* __restrict__ er, int Ncnt)
{
    int gw   = (int)((blockIdx.x * blockDim.x + threadIdx.x) >> 5);
    int lane = threadIdx.x & 31;
    if (gw >= Ncnt * 4) return;
    int n = gw >> 2, h = gw & 3;
    const float* f  = feat + (size_t)n * 512 + h * 128;
    const float* al = attn_l + h * 128;
    const float* ar = attn_r + h * 128;
    float sl = 0.f, sr = 0.f;
#pragma unroll
    for (int i = lane; i < 128; i += 32) { float v = f[i]; sl += v * al[i]; sr += v * ar[i]; }
#pragma unroll
    for (int off = 16; off; off >>= 1) { sl += __shfl_down(sl, off); sr += __shfl_down(sr, off); }
    if (lane == 0) { el[gw] = sl; er[gw] = sr; }
}

// ------------------------------ edge segment-max ------------------------------
__global__ __launch_bounds__(256) void edge_max_kernel(
    const int* __restrict__ src, const int* __restrict__ dst,
    const float4* __restrict__ el4, const float4* __restrict__ er4,
    unsigned* __restrict__ mkey, int E)
{
    int e = blockIdx.x * blockDim.x + threadIdx.x;
    if (e >= E) return;
    int s = src[e], d = dst[e];
    float4 a = el4[s], b = er4[d];
    float v[4] = { a.x + b.x, a.y + b.y, a.z + b.z, a.w + b.w };
#pragma unroll
    for (int h = 0; h < 4; ++h) {
        float t = v[h] > 0.f ? v[h] : SLOPE * v[h];
        atomicMax(&mkey[d * 4 + h], fkey(t));
    }
}

// ----------------------------- edge exp + denom -----------------------------
__global__ __launch_bounds__(256) void edge_exp_kernel(
    const int* __restrict__ src, const int* __restrict__ dst,
    const float4* __restrict__ el4, const float4* __restrict__ er4,
    const unsigned* __restrict__ mkey, float* __restrict__ exbuf,
    float* __restrict__ denom, int E)
{
    int e = blockIdx.x * blockDim.x + threadIdx.x;
    if (e >= E) return;
    int s = src[e], d = dst[e];
    float4 a = el4[s], b = er4[d];
    float v[4] = { a.x + b.x, a.y + b.y, a.z + b.z, a.w + b.w };
#pragma unroll
    for (int h = 0; h < 4; ++h) {
        float t  = v[h] > 0.f ? v[h] : SLOPE * v[h];
        float m  = fdec(mkey[d * 4 + h]);
        float ex = expf(t - m);
        exbuf[e * 4 + h] = ex;
        unsafeAtomicAdd(&denom[d * 4 + h], ex);
    }
}

// -------------------- message scatter: one wave per edge --------------------
__global__ __launch_bounds__(256) void edge_scatter_kernel(
    const int* __restrict__ src, const int* __restrict__ dst,
    const float* __restrict__ exbuf, const float* __restrict__ denom,
    const float* __restrict__ feat, float* __restrict__ rst, int E)
{
    int wave = threadIdx.x >> 5, lane = threadIdx.x & 31;
    int e = blockIdx.x * 8 + wave;
    if (e >= E) return;
    int s = src[e], d = dst[e];
    const float4* f4  = (const float4*)(feat + (size_t)s * 512);
    float*        r   = rst + (size_t)d * 512;
    float4        exv = ((const float4*)exbuf)[e];
    float4        dnv = ((const float4*)denom)[d];
    float a[4] = { exv.x / dnv.x, exv.y / dnv.y, exv.z / dnv.z, exv.w / dnv.w };
#pragma unroll
    for (int h = 0; h < 4; ++h) {
        float4 v = f4[h * 32 + lane];
        float* rp = r + h * 128 + lane * 4;
        unsafeAtomicAdd(rp + 0, v.x * a[h]);
        unsafeAtomicAdd(rp + 1, v.y * a[h]);
        unsafeAtomicAdd(rp + 2, v.z * a[h]);
        unsafeAtomicAdd(rp + 3, v.w * a[h]);
    }
}

// ----------------- final: mean heads + ELU + blended Wd GEMM -----------------
__global__ __launch_bounds__(256) void out_kernel(
    const float* __restrict__ rst, const float* __restrict__ y, const float* __restrict__ gat_b,
    const _Float16* __restrict__ Wd0h, const _Float16* __restrict__ Wd1h,
    const float* __restrict__ Wd0_b, const float* __restrict__ Wd1_b,
    float* __restrict__ out, int Ncnt)
{
    __shared__ _Float16 sH[128][136];
    __shared__ float    sP[128];

    const int n0  = blockIdx.x * 128;
    const int tid = threadIdx.x;

    for (int idx = tid; idx < 128 * 128; idx += 256) {
        int r = idx >> 7, c = idx & 127;
        int n = n0 + r;
        float v = 0.f, p = 0.f;
        if (n < Ncnt) {
            p = y[n];
            const float* rp = rst + (size_t)n * 512 + c;
            float s  = rp[0] + rp[128] + rp[256] + rp[384];
            float gb = gat_b[c] + gat_b[128 + c] + gat_b[256 + c] + gat_b[384 + c];
            float hm = (s + gb) * 0.25f;
            v = hm > 0.f ? hm : (expf(hm) - 1.f);   // ELU
        }
        if (c == 0) sP[r] = p;
        sH[r][c] = (_Float16)v;
    }
    __syncthreads();

    const int wave = tid >> 5, lane = tid & 31;
    const int r0   = wave * 16;
    const int rlo  = r0 + ((lane & 16) ? 8 : 0);
    const int cl   = lane & 15;

    for (int c = 0; c < 8; ++c) {
        v8f a0 = {}, a1 = {};
#pragma unroll
        for (int k = 0; k < 4; ++k) {
            v16h af = load_fragA(&sH[r0][k * 32], 136, lane);
            v16h b0 = load_fragB(Wd0h + (size_t)(c * 16) * 128 + k * 32, 128, lane);
            v16h b1 = load_fragB(Wd1h + (size_t)(c * 16) * 128 + k * 32, 128, lane);
            a0 = wmma_f16(af, b0, a0);
            a1 = wmma_f16(af, b1, a1);
        }
        int cn = c * 16 + cl;
        float bb0 = Wd0_b[cn], bb1 = Wd1_b[cn];
#pragma unroll
        for (int r = 0; r < 8; ++r) {
            int n = n0 + rlo + r;
            if (n < Ncnt) {
                float p = sP[rlo + r];
                out[(size_t)n * 128 + cn] = (1.f - p) * (a0[r] + bb0) + p * (a1[r] + bb1);
            }
        }
    }
}

// --------------------------------- launcher ---------------------------------
extern "C" void kernel_launch(void* const* d_in, const int* in_sizes, int n_in,
                              void* d_out, int out_size, void* d_ws, size_t ws_size,
                              hipStream_t stream) {
    const float* x      = (const float*)d_in[0];
    const float* y      = (const float*)d_in[1];
    const int*   src    = (const int*)d_in[2];
    const int*   dst    = (const int*)d_in[3];
    const float* emb    = (const float*)d_in[4];
    const float* Ws0_w  = (const float*)d_in[5];
    const float* Ws0_b  = (const float*)d_in[6];
    const float* Ws1_w  = (const float*)d_in[7];
    const float* Ws1_b  = (const float*)d_in[8];
    const float* fc_w   = (const float*)d_in[9];
    const float* attn_l = (const float*)d_in[10];
    const float* attn_r = (const float*)d_in[11];
    const float* gat_b  = (const float*)d_in[12];
    const float* Wd0_w  = (const float*)d_in[13];
    const float* Wd0_b  = (const float*)d_in[14];
    const float* Wd1_w  = (const float*)d_in[15];
    const float* Wd1_b  = (const float*)d_in[16];
    float* out = (float*)d_out;

    char*  ws  = (char*)d_ws;
    size_t off = 0;
    auto alloc = [&](size_t bytes) -> void* {
        off = (off + 255) & ~(size_t)255;
        void* p = ws + off;
        off += bytes;
        return p;
    };

    _Float16* Ws0h  = (_Float16*)alloc(128 * 128 * 2);
    _Float16* Ws1h  = (_Float16*)alloc(128 * 128 * 2);
    _Float16* fch   = (_Float16*)alloc(512 * 128 * 2);
    _Float16* Wd0h  = (_Float16*)alloc(128 * 128 * 2);
    _Float16* Wd1h  = (_Float16*)alloc(128 * 128 * 2);
    float*    feat  = (float*)alloc((size_t)NNODES * 512 * 4);
    float*    rst   = (float*)alloc((size_t)NNODES * 512 * 4);
    float*    el    = (float*)alloc((size_t)NNODES * 4 * 4);
    float*    er    = (float*)alloc((size_t)NNODES * 4 * 4);
    unsigned* mkey  = (unsigned*)alloc((size_t)NNODES * 4 * 4);
    float*    denom = (float*)alloc((size_t)NNODES * 4 * 4);
    float*    exbuf = (float*)alloc((size_t)NEDGES * 4 * 4);

    // weight conversion
    cvt_f16_kernel<<<(128 * 128 + 255) / 256, 256, 0, stream>>>(Ws0_w, Ws0h, 128 * 128);
    cvt_f16_kernel<<<(128 * 128 + 255) / 256, 256, 0, stream>>>(Ws1_w, Ws1h, 128 * 128);
    cvt_f16_kernel<<<(512 * 128 + 255) / 256, 256, 0, stream>>>(fc_w, fch, 512 * 128);
    cvt_f16_kernel<<<(128 * 128 + 255) / 256, 256, 0, stream>>>(Wd0_w, Wd0h, 128 * 128);
    cvt_f16_kernel<<<(128 * 128 + 255) / 256, 256, 0, stream>>>(Wd1_w, Wd1h, 128 * 128);

    // accumulator clears (key 0 == below all reals)
    hipMemsetAsync(mkey, 0, (size_t)NNODES * 4 * 4, stream);
    hipMemsetAsync(denom, 0, (size_t)NNODES * 4 * 4, stream);
    hipMemsetAsync(rst, 0, (size_t)NNODES * 512 * 4, stream);

    const int nodeBlocks = (NNODES + 127) / 128;
    node_feat_kernel<<<nodeBlocks, 256, 0, stream>>>(x, y, emb, Ws0_b, Ws1_b,
                                                     Ws0h, Ws1h, fch, feat, NNODES);

    attn_dot_kernel<<<(NNODES * 4 + 7) / 8, 256, 0, stream>>>(feat, attn_l, attn_r, el, er, NNODES);

    edge_max_kernel<<<(NEDGES + 255) / 256, 256, 0, stream>>>(src, dst, (const float4*)el,
                                                              (const float4*)er, mkey, NEDGES);

    edge_exp_kernel<<<(NEDGES + 255) / 256, 256, 0, stream>>>(src, dst, (const float4*)el,
                                                              (const float4*)er, mkey, exbuf,
                                                              denom, NEDGES);

    edge_scatter_kernel<<<(NEDGES + 7) / 8, 256, 0, stream>>>(src, dst, exbuf, denom,
                                                              feat, rst, NEDGES);

    out_kernel<<<nodeBlocks, 256, 0, stream>>>(rst, y, gat_b, Wd0h, Wd1h,
                                               Wd0_b, Wd1_b, out, NNODES);

    (void)in_sizes; (void)n_in; (void)out_size; (void)ws_size;
}
